// gwnetgat_8821862826031
// MI455X (gfx1250) — matrix operand order repaired and checked by reading, not compiled, not run on hardware
//
#include <hip/hip_runtime.h>
#include <hip/hip_bf16.h>

typedef __attribute__((ext_vector_type(16))) _Float16 v16h;
typedef __attribute__((ext_vector_type(2)))  _Float16 h2;
typedef __attribute__((ext_vector_type(8)))  float    v8f;

#define B_   8
#define C_   64
#define T_   24
#define N_   512
#define H_   2
#define HD_  32
#define BT_  (B_*T_)
#define LEAK 0.2f

// workspace layout in 4-byte units
#define WS_BITS 0              // 512*17 u32 (bit-packed gso, padded stride 17)
#define WS_WQE  (N_*17)        // 2*64 floats (a_src^T Wq), then 2*64 (a_dst^T Wk)
#define WS_WKE  (WS_WQE + 128)

// LDS layout (bytes)
#define VT_STRIDE   520        // halfs per row of transposed-V (padded: conflict-free)
#define BITS_STRIDE 17
#define OUT_STRIDE  68
#define L_VT    0
#define L_E     (64*VT_STRIDE*2)            // 66560   (f32 src-exp table [n][4])
#define L_BITS  (L_E + N_*4*4)              // 74752
#define L_OUT   (L_BITS + N_*BITS_STRIDE*4) // 109568
#define L_WEFF  (L_OUT + N_*OUT_STRIDE*4)   // 248832
#define L_ED    (L_WEFF + 256*4)            // 249856  (f16 dst-exp SoA [h*2+which][n])
#define LDS_BYTES (L_ED + 4*N_*2)           // 253952

// 16-bit A-matrix (16x32) K index for vector element kk (contiguous: k = 16*half + kk, matches B)
__device__ __forceinline__ int kmapA(int kk, int lane) {
  return kk + ((lane & 16) ? 16 : 0);
}

__global__ void prep_weff(const float* __restrict__ Wq, const float* __restrict__ Wk,
                          const float* __restrict__ a_src, const float* __restrict__ a_dst,
                          float* __restrict__ ws) {
  int tI = threadIdx.x;            // 0..127
  int h = tI >> 6, c = tI & 63;
  float sq = 0.f, sk = 0.f;
  for (int d = 0; d < HD_; ++d) {
    sq += a_src[h*HD_ + d] * Wq[(h*HD_ + d)*C_ + c];
    sk += a_dst[h*HD_ + d] * Wk[(h*HD_ + d)*C_ + c];
  }
  ws[WS_WQE + h*C_ + c] = sq;
  ws[WS_WKE + h*C_ + c] = sk;
}

__global__ void prep_bits(const int* __restrict__ gso, unsigned* __restrict__ bits) {
  int i = blockIdx.x;              // row
  int w = threadIdx.x;             // 0..31
  if (w >= BITS_STRIDE) return;
  unsigned v = 0;
  if (w < 16) {
    for (int bb = 0; bb < 32; ++bb)
      v |= (gso[i*N_ + w*32 + bb] != 0 ? 1u : 0u) << bb;
  }
  bits[i*BITS_STRIDE + w] = v;
}

__global__ __launch_bounds__(512, 1)
void gat_fused(const float* __restrict__ x, const float* __restrict__ Wv,
               const float* __restrict__ ws_f, const unsigned* __restrict__ ws_bits,
               const float* __restrict__ gamma, const float* __restrict__ beta,
               float* __restrict__ out) {
  extern __shared__ char smem[];
  _Float16* VT   = (_Float16*)(smem + L_VT);   // [c=64][j=512+pad] f16, V transposed
  float*    E    = (float*)(smem + L_E);       // [n][4]: Es1h0,Es2h0,Es1h1,Es2h1 (src factors, f32)
  unsigned* BITS = (unsigned*)(smem + L_BITS); // [i][17]
  float*    OUTS = (float*)(smem + L_OUT);     // [n][68] LN staging
  float*    WEFF = (float*)(smem + L_WEFF);    // 256: wqe(h0,h1), wke(h0,h1)
  _Float16* EDST = (_Float16*)(smem + L_ED);   // [(h*2+which)][n] dst factors, f16 SoA

  const int bt = blockIdx.x;
  const int b = bt / T_, t = bt % T_;
  const int tid  = threadIdx.x;
  const int lane = tid & 31;
  const int wave = tid >> 5;
  const int hlo  = lane & 15;
  const int hhi  = (lane & 16) ? 1 : 0;

  // ---- stage gso bitmask + effective score weights ----
  for (int idx = tid; idx < N_*BITS_STRIDE; idx += 512) BITS[idx] = ws_bits[idx];
  if (tid < 256) WEFF[tid] = ws_f[WS_WQE + tid];
  __syncthreads();

  // ---- per-node scores -> separable exp factors (kills 100M pairwise exps) ----
  {
    const int n = tid;
    float sq0 = 0.f, sq1 = 0.f, sk0 = 0.f, sk1 = 0.f;
    for (int c = 0; c < C_; ++c) {
      float xv = x[(((size_t)b*C_ + c)*T_ + t)*N_ + n];
      sq0 += xv * WEFF[c];
      sq1 += xv * WEFF[64 + c];
      sk0 += xv * WEFF[128 + c];
      sk1 += xv * WEFF[192 + c];
    }
    E[n*4 + 0] = expf(sq0); E[n*4 + 1] = expf(LEAK*sq0);
    E[n*4 + 2] = expf(sq1); E[n*4 + 3] = expf(LEAK*sq1);
    EDST[0*N_ + n] = (_Float16)expf(sk0);
    EDST[1*N_ + n] = (_Float16)expf(LEAK*sk0);
    EDST[2*N_ + n] = (_Float16)expf(sk1);
    EDST[3*N_ + n] = (_Float16)expf(LEAK*sk1);
  }

  // ---- V projection via WMMA: V = xr @ Wv^T, stored transposed (f16) in LDS ----
  v16h bf[4][2];
  #pragma unroll
  for (int dt = 0; dt < 4; ++dt)
    #pragma unroll
    for (int kc = 0; kc < 2; ++kc) {
      const float* wrow = Wv + (dt*16 + hlo)*C_ + kc*32 + (hhi ? 16 : 0);
      #pragma unroll
      for (int kk = 0; kk < 16; ++kk) bf[dt][kc][kk] = (_Float16)wrow[kk];
    }
  for (int nt = wave; nt < 32; nt += 16) {
    const int nrow = nt*16 + hlo;
    v16h a0, a1;
    #pragma unroll
    for (int kk = 0; kk < 16; ++kk) {
      int c0 = kmapA(kk, lane);
      a0[kk] = (_Float16)x[(((size_t)b*C_ + c0)*T_ + t)*N_ + nrow];
      a1[kk] = (_Float16)x[(((size_t)b*C_ + c0 + 32)*T_ + t)*N_ + nrow];
    }
    #pragma unroll
    for (int dt = 0; dt < 4; ++dt) {
      v8f acc = {};
      acc = __builtin_amdgcn_wmma_f32_16x16x32_f16(false, a0, false, bf[dt][0], (short)0, acc, false, false);
      acc = __builtin_amdgcn_wmma_f32_16x16x32_f16(false, a1, false, bf[dt][1], (short)0, acc, false, false);
      const int c_out = dt*16 + hlo;
      const int jbase = nt*16 + (hhi ? 8 : 0);
      union { _Float16 h[8]; uint4 u; } tmpu;
      #pragma unroll
      for (int r = 0; r < 8; ++r) tmpu.h[r] = (_Float16)acc[r];
      *(uint4*)(&VT[c_out*VT_STRIDE + jbase]) = tmpu.u;   // contiguous in j: one b128
    }
  }

  // constant all-ones B fragment: row-sum WMMA puts softmax denominators on the matrix pipe
  v16h ones;
  #pragma unroll
  for (int kk = 0; kk < 16; ++kk) ones[kk] = (_Float16)1.0f;
  __syncthreads();

  // ---- attention: P generated packed-f16 in A-fragment registers, attn@V via WMMA ----
  for (int task = wave; task < 64; task += 16) {
    const int it = task >> 1, h = task & 1;
    const int irow = it*16 + hlo;
    const float es1 = E[irow*4 + h*2], es2 = E[irow*4 + h*2 + 1];
    const float Mi  = fmaxf(es1, es2);
    const _Float16 e1 = (_Float16)(es1 / Mi), e2 = (_Float16)(es2 / Mi); // per-row scale cancels
    const h2 ei1 = { e1, e1 };
    const h2 ei2 = { e2, e2 };
    const _Float16 h1c = (_Float16)1.0f, h0c = (_Float16)0.0f;
    v8f acc0 = {}, acc1 = {}, accS = {};
    const _Float16* ED1 = EDST + (h*2 + 0)*N_;
    const _Float16* ED2 = EDST + (h*2 + 1)*N_;
    for (int j0 = 0; j0 < N_; j0 += 32) {
      const unsigned wbits = BITS[irow*BITS_STRIDE + (j0 >> 5)];
      v16h a;
      #pragma unroll
      for (int kk = 0; kk < 16; kk += 2) {
        const int jj = j0 + kmapA(kk, lane);           // even: pairs are K-contiguous
        h2 ej1 = *(const h2*)(&ED1[jj]);
        h2 ej2 = *(const h2*)(&ED2[jj]);
        h2 p2  = __builtin_elementwise_max(ei1*ej1, ei2*ej2);   // v_pk_mul/max_f16
        h2 m   = { ((wbits >> (jj & 31)) & 1u) ? h1c : h0c,
                   ((wbits >> ((jj + 1) & 31)) & 1u) ? h1c : h0c };
        p2 = p2 * m;                                   // mask: packed multiply by 0/1
        a[kk]     = p2.x;
        a[kk + 1] = p2.y;
      }
      const int dbase = h*32 + hlo;
      const int koff  = j0 + (hhi ? 16 : 0);
      union { uint4 u[2]; v16h v; } bb0, bb1;
      const char* p0 = (const char*)(&VT[dbase*VT_STRIDE + koff]);
      const char* p1 = (const char*)(&VT[(dbase + 16)*VT_STRIDE + koff]);
      bb0.u[0] = *(const uint4*)(p0); bb0.u[1] = *(const uint4*)(p0 + 16);
      bb1.u[0] = *(const uint4*)(p1); bb1.u[1] = *(const uint4*)(p1 + 16);
      acc0 = __builtin_amdgcn_wmma_f32_16x16x32_f16(false, a, false, bb0.v, (short)0, acc0, false, false);
      acc1 = __builtin_amdgcn_wmma_f32_16x16x32_f16(false, a, false, bb1.v, (short)0, acc1, false, false);
      accS = __builtin_amdgcn_wmma_f32_16x16x32_f16(false, a, false, ones,  (short)0, accS, false, false);
    }
    // accS[r] = row sum for row it*16 + r + roff (same row this lane's acc0/acc1[r] belong to)
    const int roff = hhi ? 8 : 0;
    #pragma unroll
    for (int r = 0; r < 8; ++r) {
      float inv = 1.0f / accS[r];
      OUTS[(it*16 + r + roff)*OUT_STRIDE + h*32 + hlo]      = acc0[r] * inv;
      OUTS[(it*16 + r + roff)*OUT_STRIDE + h*32 + 16 + hlo] = acc1[r] * inv;
    }
  }
  __syncthreads();

  // ---- LayerNorm over C + transposed coalesced store ----
  {
    const int n = tid;
    float s = 0.f, sq = 0.f;
    #pragma unroll
    for (int c = 0; c < C_; ++c) {
      float v = OUTS[n*OUT_STRIDE + c];
      s += v; sq += v*v;
    }
    const float mu   = s * (1.0f/C_);
    const float var  = sq * (1.0f/C_) - mu*mu;
    const float rstd = rsqrtf(var + 1e-5f);
    for (int c = 0; c < C_; ++c) {
      float v = (OUTS[n*OUT_STRIDE + c] - mu) * rstd * gamma[c] + beta[c];
      out[(((size_t)b*C_ + c)*T_ + t)*N_ + n] = v;
    }
  }
}

extern "C" void kernel_launch(void* const* d_in, const int* in_sizes, int n_in,
                              void* d_out, int out_size, void* d_ws, size_t ws_size,
                              hipStream_t stream) {
  const float* x     = (const float*)d_in[0];
  const int*   gso   = (const int*)d_in[1];
  const float* Wq    = (const float*)d_in[2];
  const float* Wk    = (const float*)d_in[3];
  const float* Wv    = (const float*)d_in[4];
  const float* a_src = (const float*)d_in[5];
  const float* a_dst = (const float*)d_in[6];
  const float* gam   = (const float*)d_in[7];
  const float* bet   = (const float*)d_in[8];
  float* out = (float*)d_out;
  float* ws  = (float*)d_ws;

  prep_weff<<<1, 128, 0, stream>>>(Wq, Wk, a_src, a_dst, ws);
  prep_bits<<<N_, 32, 0, stream>>>(gso, (unsigned*)d_ws + WS_BITS);

  (void)hipFuncSetAttribute(reinterpret_cast<const void*>(gat_fused),
                            hipFuncAttributeMaxDynamicSharedMemorySize, LDS_BYTES);
  gat_fused<<<BT_, 512, LDS_BYTES, stream>>>(x, Wv, ws,
                                             (const unsigned*)d_ws + WS_BITS,
                                             gam, bet, out);
}
